// XLSTM_36490042146960
// MI455X (gfx1250) — matrix operand order repaired and checked by reading, not compiled, not run on hardware
//
#include <hip/hip_runtime.h>
#include <hip/hip_bf16.h>

// ---------------------------------------------------------------------------
// sLSTM (xLSTM) forward for MI455X (gfx1250), bf16 WMMA + f32 accumulate.
//
// Per 256-step chunk:
//   1) gemm_wx:    Wx = A @ W.T + b   (A = LN(x) for layer0, h-seq for layer1)
//                  A tile staged into LDS by the Tensor Data Mover
//                  (tensor_load_to_lds + s_wait_tensorcnt), WMMA bf16 compute.
//   2) slstm_rec:  sequential recurrence, ONE persistent 1024-thread WG.
//                  per step: gates = Wx[t] + h @ R.T (WMMA, 32 waves x 8 tiles)
//                  -> LDS gate buffer -> elementwise i/f/o/z update in f32.
//                  h,c,n + gates live entirely in the 320KB WGP LDS.
//                  R stream prefetched with global_prefetch_b8.
// ---------------------------------------------------------------------------

typedef __bf16 bf16_t;
typedef __attribute__((ext_vector_type(16))) __bf16 v16bf;
typedef __attribute__((ext_vector_type(8)))  __bf16 v8bf;
typedef __attribute__((ext_vector_type(8)))  float  v8f;
typedef __attribute__((ext_vector_type(4)))  unsigned int v4u;
typedef __attribute__((ext_vector_type(8)))  int v8i;
typedef __attribute__((ext_vector_type(4)))  int v4i;

union BF16x16 { v16bf v; v8bf h[2]; };

#define B_SZ   8
#define L_SZ   2048
#define D_SZ   1024
#define H_SZ   1024
#define NG_SZ  4096          // 4*H
#define T_CH   256           // timesteps per chunk
#define A_PAD  1032          // rec-kernel h tile: 1024 + 8 bf16 end-of-row pad
#define GROW_B 2080          // gemm A-tile row stride in bytes (TDM-padded)

#if __has_builtin(__builtin_amdgcn_tensor_load_to_lds)
#define HAVE_TDM 1
#endif

__device__ __forceinline__ bf16_t f2bf(float f) {
    union { float f; unsigned u; } a; a.f = f;
    unsigned u = a.u;
    unsigned r = (u + 0x7FFFu + ((u >> 16) & 1u)) >> 16;  // RNE
    union { unsigned short s; bf16_t b; } o; o.s = (unsigned short)r;
    return o.b;
}

// ---------------------------------------------------------------------------
// LayerNorm: one 256-thread block per (b,l) row of 1024.
// ---------------------------------------------------------------------------
__global__ __launch_bounds__(256) void ln_kernel(
    const float* __restrict__ x, const float* __restrict__ g,
    const float* __restrict__ bb, bf16_t* __restrict__ xn)
{
    __shared__ float s1[256], s2[256];
    const long row = blockIdx.x;
    const float* xr = x + row * D_SZ;
    float v[4], a = 0.f, q = 0.f;
    #pragma unroll
    for (int i = 0; i < 4; ++i) {
        v[i] = xr[threadIdx.x + i * 256];
        a += v[i]; q += v[i] * v[i];
    }
    s1[threadIdx.x] = a; s2[threadIdx.x] = q;
    __syncthreads();
    for (int off = 128; off > 0; off >>= 1) {
        if ((int)threadIdx.x < off) {
            s1[threadIdx.x] += s1[threadIdx.x + off];
            s2[threadIdx.x] += s2[threadIdx.x + off];
        }
        __syncthreads();
    }
    const float mu   = s1[0] * (1.f / D_SZ);
    const float var  = s2[0] * (1.f / D_SZ) - mu * mu;
    const float rstd = rsqrtf(var + 1e-5f);
    #pragma unroll
    for (int i = 0; i < 4; ++i) {
        int c = threadIdx.x + i * 256;
        xn[row * D_SZ + c] = f2bf((v[i] - mu) * rstd * g[c] + bb[c]);
    }
}

// ---------------------------------------------------------------------------
// f32 -> bf16 weight conversion
// ---------------------------------------------------------------------------
__global__ __launch_bounds__(256) void cvt_kernel(
    const float* __restrict__ s, bf16_t* __restrict__ d, int n)
{
    int i = blockIdx.x * 256 + threadIdx.x;
    if (i < n) d[i] = f2bf(s[i]);
}

// ---------------------------------------------------------------------------
// init h=0, c=0, n=1 for both layers (2 * 8 * 1024 each)
// ---------------------------------------------------------------------------
__global__ __launch_bounds__(256) void init_states_kernel(
    bf16_t* __restrict__ h, float* __restrict__ c, float* __restrict__ n)
{
    int i = blockIdx.x * 256 + threadIdx.x;
    if (i < 2 * B_SZ * H_SZ) { h[i] = f2bf(0.f); c[i] = 0.f; n[i] = 1.f; }
}

// ---------------------------------------------------------------------------
// Wx = A @ W.T + bias  for one chunk of T_CH timesteps.
//   A: bf16, rows (b, t) at A + b*strideB + t*1024 (A pre-offset to chunk)
//   W: bf16 (4096, 1024) row-major  -> B-matrix column n is W row n (contig)
//   Wx: f32, row (b*T_CH + t) * 4096
// A tile (16 x 1024 bf16) is DMA'd to LDS by the TDM with hardware row
// padding: pad_interval = 256 DWORDs (1024B), pad_amount = 4 DWORDs (16B)
// -> row stride 2080B. Fragment reads (8-elem groups at multiples of 8)
// never straddle a pad point and stay 16B aligned:
//   byte(e) = row*2080 + e*2 + ((e>>9)<<4)
// ---------------------------------------------------------------------------
__global__ __launch_bounds__(256) void gemm_wx(
    const bf16_t* __restrict__ A, int strideB,
    const bf16_t* __restrict__ W, const float* __restrict__ bias,
    float* __restrict__ Wx)
{
    __shared__ __align__(16) char atile[16 * GROW_B];   // 33,280 bytes

    const int tid = threadIdx.x;
    const int tilesPerB = T_CH >> 4;                    // 16
    const int b   = blockIdx.x / tilesPerB;
    const int tl0 = (blockIdx.x % tilesPerB) << 4;
    const bf16_t* Ab = A + (long)b * strideB + (long)tl0 * D_SZ;

#ifdef HAVE_TDM
    if (tid == 0) {
        const unsigned lds_off = (unsigned)(size_t)(void*)atile;
        const unsigned long long ga = (unsigned long long)(size_t)Ab;
        // D# group 0: count=1 | lds_addr | global_addr | type=2
        v4u g0 = { 1u,
                   lds_off,
                   (unsigned)(ga & 0xFFFFFFFFu),
                   (unsigned)((ga >> 32) & 0x01FFFFFFu) | (2u << 30) };
        // D# group 1: data_size=2B(code1), pad_enable, pad_interval=256dw(7),
        // pad_amount=4dw(code3); tensor_dim0=1024, tensor_dim1=16,
        // tile_dim0=1024, tile_dim1=16, tensor_dim0_stride=1024.
        v8i g1 = { (int)((1u << 16) | (1u << 20) | (7u << 22) | (3u << 25)),
                   (int)(1024u << 16),   // [31:16] tensor_dim0 lo16
                   (int)(16u << 16),     // [15:0] dim0 hi16=0, [31:16] tensor_dim1 lo16
                   (int)(1024u << 16),   // [15:0] dim1 hi16=0, [31:16] tile_dim0
                   (int)16,              // [15:0] tile_dim1, [31:16] tile_dim2=0
                   (int)1024,            // tensor_dim0_stride lo32
                   0, 0 };
        v4i g2 = { 0, 0, 0, 0 };
        v4i g3 = { 0, 0, 0, 0 };
#if defined(__clang_major__) && (__clang_major__ >= 23)
        v8i g4 = { 0, 0, 0, 0, 0, 0, 0, 0 };
        __builtin_amdgcn_tensor_load_to_lds(g0, g1, g2, g3, g4, 0);
#else
        __builtin_amdgcn_tensor_load_to_lds(g0, g1, g2, g3, 0);
#endif
        __builtin_amdgcn_s_wait_tensorcnt(0);
    }
#else
    // fallback: cooperative load into the same padded layout
    for (int i = 0; i < 8; ++i) {
        int cidx = tid + i * 256;           // 2048 groups of 8 elems
        int row  = cidx >> 7;
        int col8 = cidx & 127;
        *(v8bf*)(atile + row * GROW_B + col8 * 16 + ((col8 >> 6) << 4)) =
            *(const v8bf*)(Ab + row * D_SZ + col8 * 8);
    }
#endif
    __syncthreads();

    const int wv    = tid >> 5;
    const int lane  = tid & 31;
    const int nlane = lane & 15;     // A row index AND B column index for this lane
    const int kh    = lane >> 4;

    v8f zero = {};
    v8f acc[4] = { zero, zero, zero, zero };

    const int nbase = blockIdx.y * 512 + wv * 64;
    const bf16_t* Wl = W + (long)(nbase + nlane) * D_SZ + kh * 16;
    const char* abase = atile + nlane * GROW_B;

    for (int kb = 0; kb < 32; ++kb) {
        BF16x16 af;
        const int e1 = kb * 32 + kh * 16;
        const int e2 = e1 + 8;
        af.h[0] = *(const v8bf*)(abase + e1 * 2 + ((e1 >> 9) << 4));
        af.h[1] = *(const v8bf*)(abase + e2 * 2 + ((e2 >> 9) << 4));
        const bf16_t* wp = Wl + kb * 32;
        #pragma unroll
        for (int j = 0; j < 4; ++j) {
            BF16x16 bw;
            bw.h[0] = *(const v8bf*)(wp + (long)j * 16 * D_SZ);
            bw.h[1] = *(const v8bf*)(wp + (long)j * 16 * D_SZ + 8);
            acc[j] = __builtin_amdgcn_wmma_f32_16x16x32_bf16(
                false, af.v, false, bw.v, (short)0, acc[j], false, false);
        }
    }

    // epilogue: add bias, store f32
    const int mbase = kh * 8;
    #pragma unroll
    for (int j = 0; j < 4; ++j) {
        int n = nbase + j * 16 + nlane;
        float bs = bias[n];
        float* outp = Wx + (long)(b * T_CH + tl0 + mbase) * NG_SZ + n;
        #pragma unroll
        for (int r = 0; r < 8; ++r)
            outp[(long)r * NG_SZ] = acc[j][r] + bs;
    }
}

// ---------------------------------------------------------------------------
// Sequential recurrence for one chunk, one layer. Single persistent WG
// (1024 thr = 32 waves). States + gate buffer in LDS (~225 KB of 320 KB WGP).
//   gates = Wx[t] + h @ R.T ; i=exp(min(.,20)) f=exp(min(.,20)) o=sig z=tanh
//   c' = f*c + i*z ; n' = f*n + i ; h' = o * c'/(n'+eps)
// mode 0: write h' (bf16) to hseq (feeds layer-1 GEMM)
// mode 1: write out = resid + h' (f32) to d_out
// ---------------------------------------------------------------------------
__global__ __launch_bounds__(1024) void slstm_rec(
    const bf16_t* __restrict__ R, const float* __restrict__ Wx,
    bf16_t* __restrict__ gh, float* __restrict__ gc, float* __restrict__ gn,
    bf16_t* __restrict__ hseq,
    float* __restrict__ out, const float* __restrict__ resid,
    int t0, int mode)
{
    extern __shared__ char smem[];
    bf16_t* h_a = (bf16_t*)smem;                               // 16 x A_PAD bf16 = 33,024 B
    float*  gts = (float*)(smem + 33024);                      // 8 x 4096 f32   = 131,072 B
    float*  cs  = (float*)(smem + 33024 + 131072);             // 8 x 1024 f32   = 32,768 B
    float*  ns  = (float*)(smem + 33024 + 131072 + 32768);     // 8 x 1024 f32   = 32,768 B

    const int tid = threadIdx.x;

    // zero-pad A tile (rows 8..15 stay zero: B=8 < tile M=16)
    for (int i = tid; i < 16 * A_PAD; i += 1024) h_a[i] = f2bf(0.f);
    __syncthreads();
    for (int i = tid; i < B_SZ * H_SZ; i += 1024) {
        int r = i >> 10, c = i & 1023;
        h_a[r * A_PAD + c] = gh[i];
        cs[i] = gc[i];
        ns[i] = gn[i];
    }
    __syncthreads();

    const int wv    = tid >> 5;
    const int lane  = tid & 31;
    const int nlane = lane & 15;
    const int kh    = lane >> 4;
    const bf16_t* Rl = R + (long)(wv * 128 + nlane) * H_SZ + kh * 16;

    const int eb  = tid >> 7;             // batch row for elementwise phase
    const int ejb = (tid & 127) << 3;     // 8 consecutive j per thread

    for (int t = 0; t < T_CH; ++t) {
        // ---- phase 1: gate strip GEMM (wave wv -> cols [wv*128, wv*128+128))
        v8f zero = {};
        v8f acc[8] = { zero, zero, zero, zero, zero, zero, zero, zero };
        for (int kb = 0; kb < 32; ++kb) {
            BF16x16 af;
            af.h[0] = *(const v8bf*)(h_a + nlane * A_PAD + kb * 32 + kh * 16);
            af.h[1] = *(const v8bf*)(h_a + nlane * A_PAD + kb * 32 + 8 + kh * 16);
            const bf16_t* rp = Rl + kb * 32;
            // pull the R stream ahead of the WMMA consumer (global_prefetch_b8)
            __builtin_prefetch((const void*)(rp + 256), 0, 0);
            #pragma unroll
            for (int j = 0; j < 8; ++j) {
                BF16x16 bw;
                bw.h[0] = *(const v8bf*)(rp + (long)j * 16 * H_SZ);
                bw.h[1] = *(const v8bf*)(rp + (long)j * 16 * H_SZ + 8);
                acc[j] = __builtin_amdgcn_wmma_f32_16x16x32_bf16(
                    false, af.v, false, bw.v, (short)0, acc[j], false, false);
            }
        }
        // store valid rows (m = 0..7 -> kh==0 lanes) to LDS gate buffer
        if (kh == 0) {
            #pragma unroll
            for (int j = 0; j < 8; ++j) {
                float* gp = gts + (wv * 128 + j * 16 + nlane);
                #pragma unroll
                for (int r = 0; r < 8; ++r) gp[r * NG_SZ] = acc[j][r];
            }
        }
        __syncthreads();

        // ---- phase 2: elementwise sLSTM update (f32), 8 elems per thread
        {
            const float* wxp  = Wx + (long)(eb * T_CH + t) * NG_SZ + ejb;
            const float* grow = gts + eb * NG_SZ + ejb;
            #pragma unroll
            for (int i2 = 0; i2 < 8; ++i2) {
                int j = ejb + i2;
                float ig = grow[i2]        + wxp[i2];
                float fg = grow[i2 + 1024] + wxp[i2 + 1024];
                float og = grow[i2 + 2048] + wxp[i2 + 2048];
                float zg = grow[i2 + 3072] + wxp[i2 + 3072];
                float iG = __expf(fminf(ig, 20.f));
                float fG = __expf(fminf(fg, 20.f));
                float oG = 1.f / (1.f + __expf(-og));
                float zG = tanhf(zg);
                int sidx = eb * H_SZ + j;
                float cN = fG * cs[sidx] + iG * zG;
                float nN = fG * ns[sidx] + iG;
                float hN = oG * (cN / (nN + 1e-8f));
                cs[sidx] = cN; ns[sidx] = nN;
                h_a[eb * A_PAD + j] = f2bf(hN);
                if (mode == 0) {
                    hseq[(long)(eb * T_CH + t) * H_SZ + j] = f2bf(hN);
                } else {
                    long oi = ((long)eb * L_SZ + (t0 + t)) * H_SZ + j;
                    out[oi] = resid[oi] + hN;
                }
            }
        }
        __syncthreads();
    }

    // save states for next chunk
    for (int i = tid; i < B_SZ * H_SZ; i += 1024) {
        int r = i >> 10, c = i & 1023;
        gh[i] = h_a[r * A_PAD + c];
        gc[i] = cs[i];
        gn[i] = ns[i];
    }
}

// ---------------------------------------------------------------------------
// Host launcher. Inputs: x, W_w, W_b, R_w, ln_g, ln_b (all f32).
// Workspace layout (~105 MB, all offsets 256B-aligned).
// ---------------------------------------------------------------------------
extern "C" void kernel_launch(void* const* d_in, const int* in_sizes, int n_in,
                              void* d_out, int out_size, void* d_ws, size_t ws_size,
                              hipStream_t stream) {
    (void)in_sizes; (void)n_in; (void)out_size; (void)ws_size;

    const float* x   = (const float*)d_in[0];
    const float* Ww  = (const float*)d_in[1];   // (2, 4096, 1024)
    const float* Wb  = (const float*)d_in[2];   // (2, 4096)
    const float* Rw  = (const float*)d_in[3];   // (2, 4096, 1024)
    const float* lng = (const float*)d_in[4];
    const float* lnb = (const float*)d_in[5];
    float* out = (float*)d_out;

    char* ws = (char*)d_ws;
    const size_t o_xn = 0;                                   // 33,554,432 B
    const size_t o_W  = o_xn + (size_t)33554432;             // 16,777,216 B
    const size_t o_R  = o_W  + (size_t)16777216;             // 16,777,216 B
    const size_t o_Wx = o_R  + (size_t)16777216;             // 33,554,432 B
    const size_t o_hs = o_Wx + (size_t)33554432;             //  4,194,304 B
    const size_t o_h  = o_hs + (size_t)4194304;              //     32,768 B
    const size_t o_c  = o_h  + (size_t)32768;                //     65,536 B
    const size_t o_n  = o_c  + (size_t)65536;                //     65,536 B

    bf16_t* xn    = (bf16_t*)(ws + o_xn);
    bf16_t* Wbf   = (bf16_t*)(ws + o_W);
    bf16_t* Rbf   = (bf16_t*)(ws + o_R);
    float*  Wxbuf = (float*) (ws + o_Wx);
    bf16_t* hseq  = (bf16_t*)(ws + o_hs);
    bf16_t* gh    = (bf16_t*)(ws + o_h);
    float*  gc    = (float*) (ws + o_c);
    float*  gn    = (float*) (ws + o_n);

    const int nW = 2 * NG_SZ * D_SZ;   // 8,388,608 per tensor

    ln_kernel<<<B_SZ * L_SZ, 256, 0, stream>>>(x, lng, lnb, xn);
    cvt_kernel<<<(nW + 255) / 256, 256, 0, stream>>>(Ww, Wbf, nW);
    cvt_kernel<<<(nW + 255) / 256, 256, 0, stream>>>(Rw, Rbf, nW);
    init_states_kernel<<<(2 * B_SZ * H_SZ + 255) / 256, 256, 0, stream>>>(gh, gc, gn);

    const size_t recSmem = 33024 + 131072 + 32768 + 32768;   // 229,632 B of 320 KB WGP LDS
    (void)hipFuncSetAttribute(reinterpret_cast<const void*>(&slstm_rec),
                              hipFuncAttributeMaxDynamicSharedMemorySize,
                              (int)recSmem);

    const dim3 ggrid(B_SZ * (T_CH / 16), NG_SZ / 512);       // (128, 8)

    for (int chunk = 0; chunk < L_SZ / T_CH; ++chunk) {
        const int t0 = chunk * T_CH;
        // ---- layer 0
        gemm_wx<<<ggrid, 256, 0, stream>>>(xn + (size_t)t0 * D_SZ, L_SZ * D_SZ,
                                           Wbf, Wb, Wxbuf);
        slstm_rec<<<1, 1024, recSmem, stream>>>(Rbf, Wxbuf, gh, gc, gn,
                                                hseq, out, x, t0, /*mode=*/0);
        // ---- layer 1 (input = layer-0 hidden sequence of this chunk)
        gemm_wx<<<ggrid, 256, 0, stream>>>(hseq, T_CH * H_SZ,
                                           Wbf + (size_t)NG_SZ * D_SZ, Wb + NG_SZ, Wxbuf);
        slstm_rec<<<1, 1024, recSmem, stream>>>(Rbf + (size_t)NG_SZ * H_SZ, Wxbuf,
                                                gh + B_SZ * H_SZ, gc + B_SZ * H_SZ, gn + B_SZ * H_SZ,
                                                hseq, out, x, t0, /*mode=*/1);
    }
}
